// AlignedSTMNCell_712964571317
// MI455X (gfx1250) — matrix operand order repaired and checked by reading, not compiled, not run on hardware
//
#include <hip/hip_runtime.h>
#include <stdint.h>

// ---------------------------------------------------------------------------
// AlignedSTMNCell for MI455X (gfx1250): bf16 WMMA implicit-GEMM convs with
// double-buffered LDS staging (1 barrier / 4 WMMAs), VALU gather kernels for
// correlation / bilinear sampling, deterministic tree reductions.
// ---------------------------------------------------------------------------

typedef unsigned short u16;
typedef __attribute__((ext_vector_type(16))) __bf16 v16bf;
typedef __attribute__((ext_vector_type(8)))  float  v8f;

union Frag { uint4 q[2]; v16bf v; };

__device__ __forceinline__ u16 f2bf(float f) {       // f32 -> bf16 RNE
  unsigned u = __float_as_uint(f);
  u += 0x7FFFu + ((u >> 16) & 1u);
  return (u16)(u >> 16);
}

#define WMMA_BF16(a, b, c) \
  __builtin_amdgcn_wmma_f32_16x16x32_bf16(false, (a), false, (b), (short)0, (c), false, false)

#define CCH 256
#define HH 64
#define WW 64
#define HW 4096
#define FEAT 8388608ull  // B*C*H*W

// ---------------------------------------------------------------------------
// Weight pack: w [256][Cin][3][3] f32  ->  dst [256][9][Cin] bf16
// ---------------------------------------------------------------------------
__global__ void k_pack_w(const float* __restrict__ w, u16* __restrict__ dst, int Cin) {
  size_t i = (size_t)blockIdx.x * 256 + threadIdx.x;
  size_t total = (size_t)256 * 9 * Cin;
  if (i >= total) return;
  int c = (int)(i % Cin);
  size_t mt = i / Cin;
  int tap = (int)(mt % 9);
  int m = (int)(mt / 9);
  dst[i] = f2bf(w[((size_t)m * Cin + c) * 9 + tap]);
}

// ---------------------------------------------------------------------------
// Fused correlation (d=4, 81 disp, channel-mean) + 1x1 offset conv (81->18)
// One thread per pixel; coalesced across x within a wave. L2-resident gather.
// ---------------------------------------------------------------------------
__global__ void k_corr_offset(const float* __restrict__ x1, const float* __restrict__ x2,
                              const float* __restrict__ woff, float* __restrict__ off) {
  int idx = blockIdx.x * 256 + threadIdx.x;  // pixel id, 32768 total
  int b = idx >> 12;
  int y = (idx >> 6) & 63;
  int x = idx & 63;
  float acc[18];
#pragma unroll
  for (int o = 0; o < 18; ++o) acc[o] = 0.f;
  const float* p1 = x1 + (size_t)b * CCH * HW + y * WW + x;
  for (int dy = 0; dy < 9; ++dy) {
    int row = y + dy - 4;
    if (row < 0 || row >= HH) continue;  // zero-padded rows contribute nothing
    float corr[9];
#pragma unroll
    for (int d = 0; d < 9; ++d) corr[d] = 0.f;
    const float* p2 = x2 + (size_t)b * CCH * HW + row * WW;
    for (int c = 0; c < CCH; ++c) {
      float a = p1[(size_t)c * HW];
      const float* p2c = p2 + (size_t)c * HW;
#pragma unroll
      for (int d = 0; d < 9; ++d) {
        int xx = x + d - 4;
        corr[d] += (xx >= 0 && xx < WW) ? a * p2c[xx] : 0.f;
      }
    }
#pragma unroll
    for (int d = 0; d < 9; ++d) corr[d] *= (1.f / 256.f);  // channel mean
#pragma unroll
    for (int o = 0; o < 18; ++o) {
      float s = 0.f;
#pragma unroll
      for (int d = 0; d < 9; ++d) s += woff[o * 81 + dy * 9 + d] * corr[d];
      acc[o] += s;
    }
  }
#pragma unroll
  for (int o = 0; o < 18; ++o)
    off[(((size_t)b * 18 + o) * HH + y) * WW + x] = acc[o];
}

// ---------------------------------------------------------------------------
// Deformable 3x3 conv via WMMA. Tile: 256 out-ch x 32 pixels. 8 waves; wave w
// owns M-tiles {16w, 16(w+8)} x N-tiles {0,1} -> 4 WMMAs / iteration.
// Double-buffered S (per-tap bilinear samples) and A (weights); 1 barrier/iter.
// ---------------------------------------------------------------------------
__global__ void k_deform_wmma(const float* __restrict__ state, const float* __restrict__ off,
                              const u16* __restrict__ wA, float* __restrict__ out) {
  __shared__ u16 Sm[2][32 * 256];  // [buf][pixel][c] bf16 samples (2 x 16KB)
  __shared__ u16 Am[2][256 * 32];  // [buf][m][c] bf16 weights   (2 x 16KB)
  const int tid = threadIdx.x;
  const int lane = tid & 31;
  const int wid = tid >> 5;
  const int b = blockIdx.y >> 6;
  const int y = blockIdx.y & 63;
  const int x0 = blockIdx.x * 32;
  v8f acc00 = {}, acc01 = {}, acc10 = {}, acc11 = {};
  const int p = tid & 31;          // pixel within tile
  const int cb = (tid >> 5) * 32;  // 32-channel group
  const int xg = x0 + p;

  auto stageS = [&](int buf, int tap) {  // bilinear gather for one tap
    int ky = tap / 3, kx = tap % 3;
    float oy = off[(((size_t)b * 18 + 2 * tap) * HH + y) * WW + xg];
    float ox = off[(((size_t)b * 18 + 2 * tap + 1) * HH + y) * WW + xg];
    float sy = (float)(y + ky - 1) + oy;
    float sx = (float)(xg + kx - 1) + ox;
    float yf = floorf(sy), xf = floorf(sx);
    int iy0 = (int)yf, ix0 = (int)xf;
    float wy1 = sy - yf, wx1 = sx - xf;
    float wy0 = 1.f - wy1, wx0 = 1.f - wx1;
    bool vy0 = (iy0 >= 0) && (iy0 < HH);
    bool vy1 = (iy0 + 1 >= 0) && (iy0 + 1 < HH);
    bool vx0 = (ix0 >= 0) && (ix0 < WW);
    bool vx1 = (ix0 + 1 >= 0) && (ix0 + 1 < WW);
    int y0c = min(max(iy0, 0), HH - 1), y1c = min(max(iy0 + 1, 0), HH - 1);
    int x0c = min(max(ix0, 0), WW - 1), x1c = min(max(ix0 + 1, 0), WW - 1);
    float m00 = (vy0 && vx0) ? wy0 * wx0 : 0.f;
    float m01 = (vy0 && vx1) ? wy0 * wx1 : 0.f;
    float m10 = (vy1 && vx0) ? wy1 * wx0 : 0.f;
    float m11 = (vy1 && vx1) ? wy1 * wx1 : 0.f;
    int i00 = y0c * WW + x0c, i01 = y0c * WW + x1c;
    int i10 = y1c * WW + x0c, i11 = y1c * WW + x1c;
    const float* basep = state + ((size_t)b * CCH + cb) * HW;
#pragma unroll 4
    for (int i = 0; i < 32; ++i) {
      const float* pc = basep + (size_t)i * HW;
      float v = m00 * pc[i00] + m01 * pc[i01] + m10 * pc[i10] + m11 * pc[i11];
      Sm[buf][p * 256 + cb + i] = f2bf(v);
    }
  };
  auto stageA = [&](int buf, int tap, int cc) {  // 1024 uint4, 4/thread
    int q = tid * 4;
#pragma unroll
    for (int k = 0; k < 4; ++k) {
      int m = (q + k) >> 2;
      int part = (q + k) & 3;
      const uint4* s = ((const uint4*)(wA + ((size_t)m * 9 + tap) * 256 + cc * 32)) + part;
      ((uint4*)Am[buf])[q + k] = *s;
    }
  };

  stageS(0, 0);
  stageA(0, 0, 0);
  int t = 0;
  for (int tap = 0; tap < 9; ++tap) {
    for (int cc = 0; cc < 8; ++cc, ++t) {
      __syncthreads();
      // prefetch next stages into the other buffers (overlaps with WMMAs)
      int ccN = cc + 1, tapN = tap;
      if (ccN == 8) { ccN = 0; tapN = tap + 1; }
      if (tapN < 9) stageA((t + 1) & 1, tapN, ccN);
      if (cc == 0 && tap + 1 < 9) stageS((tap + 1) & 1, tap + 1);
      // compute from current buffers
      const u16* A = Am[t & 1];
      const u16* S = Sm[tap & 1];
      const int mr = lane & 15;
      const int hi = (lane >= 16) ? 8 : 0;
      const int kb = (lane >= 16) ? 8 : 0;
      Frag a0, a1, b0, b1;
      const u16* ar0 = A + (wid * 16 + mr) * 32;
      const u16* ar1 = A + ((wid + 8) * 16 + mr) * 32;
      a0.q[0] = *(const uint4*)(ar0 + hi);
      a0.q[1] = *(const uint4*)(ar0 + hi + 16);
      a1.q[0] = *(const uint4*)(ar1 + hi);
      a1.q[1] = *(const uint4*)(ar1 + hi + 16);
      const u16* br0 = S + (lane & 15) * 256 + cc * 32 + kb;
      const u16* br1 = S + ((lane & 15) + 16) * 256 + cc * 32 + kb;
      b0.q[0] = *(const uint4*)(br0);
      b0.q[1] = *(const uint4*)(br0 + 16);
      b1.q[0] = *(const uint4*)(br1);
      b1.q[1] = *(const uint4*)(br1 + 16);
      acc00 = WMMA_BF16(a0.v, b0.v, acc00);
      acc01 = WMMA_BF16(a0.v, b1.v, acc01);
      acc10 = WMMA_BF16(a1.v, b0.v, acc10);
      acc11 = WMMA_BF16(a1.v, b1.v, acc11);
    }
  }
  const int n = lane & 15;
  const int hi8 = (lane >= 16) ? 8 : 0;
#pragma unroll
  for (int r = 0; r < 8; ++r) {
    int ma = wid * 16 + r + hi8;
    int mb = (wid + 8) * 16 + r + hi8;
    size_t ba = (((size_t)b * CCH + ma) * HH + y) * WW;
    size_t bbq = (((size_t)b * CCH + mb) * HH + y) * WW;
    out[ba + x0 + n] = acc00[r];
    out[ba + x0 + 16 + n] = acc01[r];
    out[bbq + x0 + n] = acc10[r];
    out[bbq + x0 + 16 + n] = acc11[r];
  }
}

// ---------------------------------------------------------------------------
// 3x3 conv, Cin=512 (src0: ch 0..255, src1: ch 256..511) via WMMA.
// Tile: 256 M x 32 pixels. 8 waves; wave w owns M-tiles {16w,16(w+8)} x 2 N
// tiles -> 4 WMMAs / iteration. Double-buffered A (per tap-chunk) and B (per
// 32-ch chunk, 3 rows x 34 px so kx is a column shift). 1 barrier/iter.
// ---------------------------------------------------------------------------
__global__ void k_conv3x3_wmma(const float* __restrict__ src0, const float* __restrict__ src1,
                               const u16* __restrict__ wA,
                               const float* __restrict__ bias0, const float* __restrict__ bias1,
                               float* __restrict__ out0, float* __restrict__ out1) {
  __shared__ u16 Am[2][256 * 32];     // 2 x 16KB weights
  __shared__ u16 Bl[2][3 * 34 * 32];  // 2 x 6.4KB input rows
  const int tid = threadIdx.x;
  const int lane = tid & 31;
  const int wid = tid >> 5;
  const int b = blockIdx.y >> 6;
  const int y = blockIdx.y & 63;
  const int x0 = blockIdx.x * 32;
  const int m0 = blockIdx.z * 256;
  v8f acc00 = {}, acc01 = {}, acc10 = {}, acc11 = {};

  auto stageA = [&](int buf, int ccb, int tap) {
    int q = tid * 4;
#pragma unroll
    for (int k = 0; k < 4; ++k) {
      int m = (q + k) >> 2;
      int part = (q + k) & 3;
      const uint4* s =
          ((const uint4*)(wA + (((size_t)(m0 + m) * 9 + tap) * 512 + ccb * 32))) + part;
      ((uint4*)Am[buf])[q + k] = *s;
    }
  };
  auto stageB = [&](int buf, int ccb) {
    for (int e = tid; e < 3 * 34 * 32; e += 256) {
      int ky = e / 1088;
      int rem = e - ky * 1088;
      int px = rem >> 5;
      int cl = rem & 31;
      int row = y + ky - 1;
      int xgl = x0 - 1 + px;
      int c = ccb * 32 + cl;
      float v = 0.f;
      if (row >= 0 && row < HH && xgl >= 0 && xgl < WW) {
        const float* s = (c < 256) ? (src0 + ((size_t)b * 256 + c) * HW)
                                   : (src1 + ((size_t)b * 256 + (c - 256)) * HW);
        v = s[row * WW + xgl];
      }
      Bl[buf][e] = f2bf(v);
    }
  };

  stageB(0, 0);
  stageA(0, 0, 0);
  int t = 0;
  for (int ccb = 0; ccb < 16; ++ccb) {
    for (int tap = 0; tap < 9; ++tap, ++t) {
      __syncthreads();
      // prefetch next stages into the other buffers (overlaps with WMMAs)
      int tapN = tap + 1, ccbN = ccb;
      if (tapN == 9) { tapN = 0; ccbN = ccb + 1; }
      if (ccbN < 16) stageA((t + 1) & 1, ccbN, tapN);
      if (tap == 0 && ccb + 1 < 16) stageB((ccb + 1) & 1, ccb + 1);
      // compute from current buffers
      const u16* A = Am[t & 1];
      const u16* Bp = Bl[ccb & 1];
      int ky = tap / 3, kx = tap % 3;
      int mr = lane & 15;
      int hi = (lane >= 16) ? 8 : 0;
      int kb = (lane >= 16) ? 8 : 0;
      Frag a0, a1, b0, b1;
      const u16* ar0 = A + (wid * 16 + mr) * 32;
      const u16* ar1 = A + ((wid + 8) * 16 + mr) * 32;
      a0.q[0] = *(const uint4*)(ar0 + hi);
      a0.q[1] = *(const uint4*)(ar0 + hi + 16);
      a1.q[0] = *(const uint4*)(ar1 + hi);
      a1.q[1] = *(const uint4*)(ar1 + hi + 16);
      int px0 = (lane & 15) + kx;  // Bl pixel index: px 0 <-> x0-1
      const u16* br0 = Bp + ((ky * 34 + px0) * 32) + kb;
      const u16* br1 = Bp + ((ky * 34 + px0 + 16) * 32) + kb;
      b0.q[0] = *(const uint4*)(br0);
      b0.q[1] = *(const uint4*)(br0 + 16);
      b1.q[0] = *(const uint4*)(br1);
      b1.q[1] = *(const uint4*)(br1 + 16);
      acc00 = WMMA_BF16(a0.v, b0.v, acc00);
      acc01 = WMMA_BF16(a0.v, b1.v, acc01);
      acc10 = WMMA_BF16(a1.v, b0.v, acc10);
      acc11 = WMMA_BF16(a1.v, b1.v, acc11);
    }
  }
  const int hi8 = (lane >= 16) ? 8 : 0;
  const int n = lane & 15;
#pragma unroll
  for (int r = 0; r < 8; ++r) {
#pragma unroll
    for (int mt = 0; mt < 2; ++mt) {
      int mg = m0 + (wid + mt * 8) * 16 + r + hi8;
      float bias = (mg < 256) ? bias0[mg] : bias1[mg - 256];
      float* dst = (mg < 256) ? out0 : out1;
      int mm = (mg < 256) ? mg : (mg - 256);
      size_t base = (((size_t)b * 256 + mm) * HH + y) * WW;
      float v0 = (mt ? acc10[r] : acc00[r]) + bias;
      float v1 = (mt ? acc11[r] : acc01[r]) + bias;
      dst[base + x0 + n] = v0;
      dst[base + x0 + 16 + n] = v1;
    }
  }
}

// ---------------------------------------------------------------------------
// linear_scale stats, deterministic two-level tree reduction (no fp atomics).
// ---------------------------------------------------------------------------
__global__ void k_reduce_stats(const float* __restrict__ updPre, const float* __restrict__ rstPre,
                               double* __restrict__ part) {
  const float* src = (blockIdx.y == 0) ? updPre : rstPre;
  __shared__ double sc[256], ss[256], sq[256];
  const size_t chunk = FEAT / 1024;  // 8192
  size_t base = (size_t)blockIdx.x * chunk;
  double c = 0.0, s = 0.0, q = 0.0;
  for (size_t i = threadIdx.x; i < chunk; i += 256) {
    float v = src[base + i];
    if (v > 0.f) { c += 1.0; s += (double)v; q += (double)v * (double)v; }
  }
  sc[threadIdx.x] = c; ss[threadIdx.x] = s; sq[threadIdx.x] = q;
  __syncthreads();
  for (int st = 128; st > 0; st >>= 1) {
    if (threadIdx.x < st) {
      sc[threadIdx.x] += sc[threadIdx.x + st];
      ss[threadIdx.x] += ss[threadIdx.x + st];
      sq[threadIdx.x] += sq[threadIdx.x + st];
    }
    __syncthreads();
  }
  if (threadIdx.x == 0) {
    size_t o = ((size_t)blockIdx.y * 1024 + blockIdx.x) * 3;
    part[o] = sc[0]; part[o + 1] = ss[0]; part[o + 2] = sq[0];
  }
}

__global__ void k_stats_final(const double* __restrict__ part, float* __restrict__ ub) {
  __shared__ double sc[256], ss[256], sq[256];
  for (int set = 0; set < 2; ++set) {
    double c = 0.0, s = 0.0, q = 0.0;
    for (int r = threadIdx.x; r < 1024; r += 256) {
      size_t o = ((size_t)set * 1024 + r) * 3;
      c += part[o]; s += part[o + 1]; q += part[o + 2];
    }
    sc[threadIdx.x] = c; ss[threadIdx.x] = s; sq[threadIdx.x] = q;
    __syncthreads();
    for (int st = 128; st > 0; st >>= 1) {
      if (threadIdx.x < st) {
        sc[threadIdx.x] += sc[threadIdx.x + st];
        ss[threadIdx.x] += ss[threadIdx.x + st];
        sq[threadIdx.x] += sq[threadIdx.x + st];
      }
      __syncthreads();
    }
    if (threadIdx.x == 0) {
      double cnt = sc[0], sum = ss[0], sumsq = sq[0];
      double mean = sum / fmax(cnt, 1.0);
      double var = (sumsq - cnt * mean * mean) / fmax(cnt - 1.0, 1.0);
      var = fmax(var, 0.0);
      double u = mean + 3.0 * sqrt(var);
      ub[set] = (cnt >= 2.0) ? (float)u : 1.0f;  // ref: cnt<2 -> ub=1
    }
    __syncthreads();
  }
}

// ---------------------------------------------------------------------------
// Elementwise passes (float4 vectorized; pure bandwidth)
// ---------------------------------------------------------------------------
__global__ void k_make_hr(const float4* __restrict__ rstPre, const float4* __restrict__ state,
                          const float* __restrict__ ub, float4* __restrict__ hr) {
  size_t i = (size_t)blockIdx.x * 256 + threadIdx.x;  // FEAT/4 elements
  if (i >= FEAT / 4) return;
  float u = ub[1];
  float4 rp = rstPre[i], st = state[i], o;
  o.x = st.x * (fminf(fmaxf(rp.x, 0.f), u) / u);
  o.y = st.y * (fminf(fmaxf(rp.y, 0.f), u) / u);
  o.z = st.z * (fminf(fmaxf(rp.z, 0.f), u) / u);
  o.w = st.w * (fminf(fmaxf(rp.w, 0.f), u) / u);
  hr[i] = o;
}

__global__ void k_final(const float4* __restrict__ updPre, const float4* __restrict__ outPre,
                        const float4* __restrict__ state, const float* __restrict__ ub,
                        float4* __restrict__ out) {
  size_t i = (size_t)blockIdx.x * 256 + threadIdx.x;  // FEAT/4 elements
  if (i >= FEAT / 4) return;
  float uu = ub[0];
  float4 up = updPre[i], op = outPre[i], st = state[i], ns;
  float u0 = fminf(fmaxf(up.x, 0.f), uu) / uu;
  float u1 = fminf(fmaxf(up.y, 0.f), uu) / uu;
  float u2 = fminf(fmaxf(up.z, 0.f), uu) / uu;
  float u3 = fminf(fmaxf(up.w, 0.f), uu) / uu;
  ns.x = st.x * (1.f - u0) + fmaxf(op.x, 0.f) * u0;
  ns.y = st.y * (1.f - u1) + fmaxf(op.y, 0.f) * u1;
  ns.z = st.z * (1.f - u2) + fmaxf(op.z, 0.f) * u2;
  ns.w = st.w * (1.f - u3) + fmaxf(op.w, 0.f) * u3;
  out[i] = ns;               // out_feat
  out[i + FEAT / 4] = ns;    // new_state
}

// ---------------------------------------------------------------------------
// Workspace layout (bytes)
// ---------------------------------------------------------------------------
static constexpr size_t OFF_OFFSET  = 0;
static constexpr size_t SZ_OFFSET   = (size_t)8 * 18 * 64 * 64 * 4;   // 2,359,296
static constexpr size_t OFF_ALIGNED = OFF_OFFSET + SZ_OFFSET;
static constexpr size_t SZ_FEATB    = FEAT * 4;                       // 33,554,432
static constexpr size_t OFF_UPD     = OFF_ALIGNED + SZ_FEATB;
static constexpr size_t OFF_RST     = OFF_UPD + SZ_FEATB;
static constexpr size_t OFF_HR      = OFF_RST + SZ_FEATB;
static constexpr size_t OFF_OUTP    = OFF_HR + SZ_FEATB;
static constexpr size_t OFF_WUR     = OFF_OUTP + SZ_FEATB;
static constexpr size_t SZ_WUR      = (size_t)512 * 9 * 512 * 2;      // 4,718,592
static constexpr size_t OFF_WOUT    = OFF_WUR + SZ_WUR;
static constexpr size_t SZ_WOUT     = (size_t)256 * 9 * 512 * 2;
static constexpr size_t OFF_WAL     = OFF_WOUT + SZ_WOUT;
static constexpr size_t SZ_WAL      = (size_t)256 * 9 * 256 * 2;
static constexpr size_t OFF_PART    = OFF_WAL + SZ_WAL;
static constexpr size_t SZ_PART     = (size_t)2 * 1024 * 3 * 8;
static constexpr size_t OFF_UB      = OFF_PART + SZ_PART;

extern "C" void kernel_launch(void* const* d_in, const int* in_sizes, int n_in,
                              void* d_out, int out_size, void* d_ws, size_t ws_size,
                              hipStream_t stream) {
  (void)in_sizes; (void)n_in; (void)out_size; (void)ws_size;
  const float* inputs   = (const float*)d_in[0];
  const float* in_state = (const float*)d_in[1];
  const float* w_reset  = (const float*)d_in[2];
  const float* b_reset  = (const float*)d_in[3];
  const float* w_update = (const float*)d_in[4];
  const float* b_update = (const float*)d_in[5];
  const float* w_out    = (const float*)d_in[6];
  const float* b_out    = (const float*)d_in[7];
  const float* w_offset = (const float*)d_in[8];
  const float* w_align  = (const float*)d_in[9];

  char* ws = (char*)d_ws;
  float*  offsetBuf = (float*)(ws + OFF_OFFSET);
  float*  alignedBuf= (float*)(ws + OFF_ALIGNED);
  float*  updPre    = (float*)(ws + OFF_UPD);
  float*  rstPre    = (float*)(ws + OFF_RST);
  float*  hrBuf     = (float*)(ws + OFF_HR);
  float*  outPre    = (float*)(ws + OFF_OUTP);
  u16*    wUR       = (u16*)(ws + OFF_WUR);
  u16*    wOUT      = (u16*)(ws + OFF_WOUT);
  u16*    wAL       = (u16*)(ws + OFF_WAL);
  double* part      = (double*)(ws + OFF_PART);
  float*  ub        = (float*)(ws + OFF_UB);

  // 1) pack weights to bf16 [M][tap][Cin]
  k_pack_w<<<4608, 256, 0, stream>>>(w_update, wUR, 512);
  k_pack_w<<<4608, 256, 0, stream>>>(w_reset,  wUR + (size_t)256 * 9 * 512, 512);
  k_pack_w<<<4608, 256, 0, stream>>>(w_out,    wOUT, 512);
  k_pack_w<<<2304, 256, 0, stream>>>(w_align,  wAL, 256);

  // 2) correlation + 1x1 offset conv
  k_corr_offset<<<128, 256, 0, stream>>>(inputs, in_state, w_offset, offsetBuf);

  // 3) deformable conv (WMMA, 256M x 32N tiles)
  k_deform_wmma<<<dim3(2, 512), 256, 0, stream>>>(in_state, offsetBuf, wAL, alignedBuf);

  // 4) fused update+reset conv (M=512): src = concat(inputs, aligned)
  k_conv3x3_wmma<<<dim3(2, 512, 2), 256, 0, stream>>>(
      inputs, alignedBuf, wUR, b_update, b_reset, updPre, rstPre);

  // 5) linear_scale stats (deterministic)
  k_reduce_stats<<<dim3(1024, 2), 256, 0, stream>>>(updPre, rstPre, part);
  k_stats_final<<<1, 256, 0, stream>>>(part, ub);

  // 6) h_r = state * scaled(reset)
  k_make_hr<<<8192, 256, 0, stream>>>((const float4*)rstPre, (const float4*)in_state, ub,
                                      (float4*)hrBuf);

  // 7) out-gate conv (M=256): src = concat(inputs, h_r)
  k_conv3x3_wmma<<<dim3(2, 512, 1), 256, 0, stream>>>(
      inputs, hrBuf, wOUT, b_out, b_out, outPre, outPre);

  // 8) final GRU blend, write (out_feat, new_state)
  k_final<<<8192, 256, 0, stream>>>((const float4*)updPre, (const float4*)outPre,
                                    (const float4*)in_state, ub, (float4*)d_out);
}
